// BlockFFTDirectPrior_53291954209346
// MI455X (gfx1250) — compile-verified, matrix operand order checked by reading, not run on hardware
//
#include <hip/hip_runtime.h>
#include <hip/hip_bf16.h>
#include <math.h>

// ---------------------------------------------------------------------------
// Block-circulant layer via frequency-domain factorization on CDNA5 (gfx1250).
//   B=2048 batch, 64 in-blocks, 64 out-blocks, block=256.
//   Stage 1: Xr/Xi[f][b][j] = DFT_256(x block j)        (WMMA bf16, K=256)
//   Stage 2: Y[f] = X[f] @ conj(Wfull[f])^T per bin     (complex WMMA, K=64,
//            W tile staged in LDS via global_load_async_to_lds_b128)
//   Stage 3: out  = Yr@(cos/256) + Yi@(-sin/256)        (WMMA bf16, K=256)
// Each wave computes a 16x64 D-slab (4 accumulators) so every A fragment is
// reused 4x -- the job is HBM-bound (~260 MB mandatory @ 23.3 TB/s ~= 11 us),
// so A-fragment traffic is the term to minimize.
// ---------------------------------------------------------------------------

#define CDNA5_ASYNC_STAGE 1   // use global_load_async_to_lds_b128 in stage 2

typedef __attribute__((ext_vector_type(16))) __bf16 v16bf;
typedef __attribute__((ext_vector_type(8)))  float  v8f;

#define BATCH   2048
#define KIN     64
#define KOUT    64
#define BLK     256
#define KHALF   129
#define INF     16384   // in/out features

// ---- workspace layout (bytes) ---------------------------------------------
#define WS_COSF   ((size_t)0)                       // bf16[256][256]  cos(2pi f n/256)
#define WS_SINF   ((size_t)131072)                  // bf16[256][256] -sin(2pi f n/256)
#define WS_ICOS   ((size_t)262144)                  // bf16[256][256]  cos/256
#define WS_ISIN   ((size_t)393216)                  // bf16[256][256] -sin/256
#define WS_WR     ((size_t)524288)                  // bf16[256][64][64]  Re conj(Wfull)
#define WS_WI     ((size_t)2621440)                 // bf16[256][64][64]  Im conj(Wfull)
#define WS_XR     ((size_t)4718592)                 // bf16[256][2048][64]
#define WS_XI     ((size_t)71827456)                // bf16[256][2048][64]
#define WS_YR     ((size_t)138936320)               // bf16[64][2048][256]
#define WS_YI     ((size_t)206045184)               // bf16[64][2048][256]

// ---- fragment helpers (documented 16-bit A layout; B^T loaded mirrored) ----
// lane<16: row M=lane, holds K = kb..kb+7 and 16+kb..16+kb+7 with kb=0
// lane>=16: row M=lane-16, same with kb=8
__device__ __forceinline__ v16bf frag_bf16(const __bf16* rowp, int kk, int kb) {
  v16bf v;
  const __bf16* s = rowp + kk + kb;
#pragma unroll
  for (int t = 0; t < 8; ++t) { v[t] = s[t]; v[8 + t] = s[16 + t]; }
  return v;
}

__device__ __forceinline__ v16bf frag_f32(const float* rowp, int kk, int kb) {
  v16bf v;
  const float* s = rowp + kk + kb;
#pragma unroll
  for (int t = 0; t < 8; ++t) {
    v[t]     = (__bf16)s[t];
    v[8 + t] = (__bf16)s[16 + t];
  }
  return v;
}

__device__ __forceinline__ v8f wmma_bf16(v16bf a, v16bf b, v8f c) {
  // 8 args: (neg_a, A, neg_b, B, c_mod, C, reuse_a, reuse_b)
  return __builtin_amdgcn_wmma_f32_16x16x32_bf16(false, a, false, b, (short)0, c,
                                                 false, false);
}

// ---- kernel 0a: twiddle tables --------------------------------------------
__global__ __launch_bounds__(256) void k0_tables(__bf16* cosF, __bf16* sinF,
                                                 __bf16* icosT, __bf16* isinT) {
  int idx = blockIdx.x * 256 + threadIdx.x;      // 65536 threads: f*256+n
  int f = idx >> 8, n = idx & 255;
  int t = (f * n) & 255;                          // periodic phase index
  float ang = 6.283185307179586f * (float)t * (1.0f / 256.0f);
  float c = cosf(ang), s = sinf(ang);
  cosF[idx]  = (__bf16)c;
  sinF[idx]  = (__bf16)(-s);                      // Xi[f] = sum x[n]*(-sin)
  icosT[idx] = (__bf16)(c * (1.0f / 256.0f));     // symmetric in (f,n)
  isinT[idx] = (__bf16)(-s * (1.0f / 256.0f));
}

// ---- kernel 0b: reconstruct conj(full spectrum) of W ----------------------
// Wfull[f] = H[f] (f<=128, H[128] real) else conj(H[256-f]);  we store conj(Wfull).
__global__ __launch_bounds__(256) void k0_w(const float* Wr, const float* Wi,
                                            __bf16* WR, __bf16* WI) {
  int idx = blockIdx.x * 256 + threadIdx.x;       // 1048576 threads: f*4096 + i*64+j
  int f  = idx >> 12;
  int ij = idx & 4095;
  int hf = (f <= 128) ? f : 256 - f;
  float sgn = (f <= 128) ? -1.0f : 1.0f;          // conj of Wfull
  float wr = Wr[(size_t)ij * KHALF + hf];
  float wi = Wi[(size_t)ij * KHALF + hf];         // zero at f=0,128 per setup
  WR[((size_t)f << 12) + ij] = (__bf16)wr;        // layout [f][i][j]
  WI[((size_t)f << 12) + ij] = (__bf16)(sgn * wi);
}

// ---- kernel 1: forward DFT of every 256-sample input block ----------------
// Per wave: 16 batch rows x 64 freq bins (4 tiles, A reused 4x), K = 256.
__global__ __launch_bounds__(32) void k1_fdft(const float* __restrict__ x,
                                              const __bf16* __restrict__ cosF,
                                              const __bf16* __restrict__ sinF,
                                              __bf16* __restrict__ Xr,
                                              __bf16* __restrict__ Xi) {
  const int lane = threadIdx.x;
  const int row  = lane & 15;
  const int kb   = (lane >> 4) * 8;
  const int b0   = blockIdx.x * 16;
  const int j    = blockIdx.y;
  const int f0   = blockIdx.z * 64;               // 4 f-tiles per wave

  const float* xrow = x + (size_t)(b0 + row) * INF + j * BLK;

  v8f accR[4] = {};
  v8f accI[4] = {};
#pragma unroll
  for (int kk = 0; kk < BLK; kk += 32) {
    __builtin_prefetch(xrow + kk + 64, 0, 0);      // global_prefetch_b8
    v16bf a = frag_f32(xrow, kk, kb);              // one A per 8 WMMAs
#pragma unroll
    for (int ft = 0; ft < 4; ++ft) {
      const __bf16* crow = cosF + (size_t)(f0 + ft * 16 + row) * BLK;
      const __bf16* srow = sinF + (size_t)(f0 + ft * 16 + row) * BLK;
      accR[ft] = wmma_bf16(a, frag_bf16(crow, kk, kb), accR[ft]);
      accI[ft] = wmma_bf16(a, frag_bf16(srow, kk, kb), accI[ft]);
    }
  }

  const int m = (lane >> 4) * 8, n = lane & 15;
#pragma unroll
  for (int ft = 0; ft < 4; ++ft) {
#pragma unroll
    for (int r = 0; r < 8; ++r) {
      size_t b = (size_t)(b0 + m + r);
      size_t f = (size_t)(f0 + ft * 16 + n);
      size_t o = (f * BATCH + b) * KIN + j;        // [f][b][j]
      Xr[o] = (__bf16)accR[ft][r];
      Xi[o] = (__bf16)accI[ft][r];
    }
  }
}

// ---- kernel 2: per-frequency complex GEMM Y = X @ conjW^T -----------------
// Workgroup = 8 waves sharing one bin f: WR[f]/WI[f] (16 KB) staged in LDS via
// async-to-LDS, then each wave computes 16 batch x all 64 out-blocks (K=64).
__global__ __launch_bounds__(256) void k2_fgemm(const __bf16* __restrict__ Xr,
                                               const __bf16* __restrict__ Xi,
                                               const __bf16* __restrict__ WR,
                                               const __bf16* __restrict__ WI,
                                               __bf16* __restrict__ Yr,
                                               __bf16* __restrict__ Yi) {
  __shared__ __align__(16) __bf16 lwr[KOUT * KIN];   // 8 KB
  __shared__ __align__(16) __bf16 lwi[KOUT * KIN];   // 8 KB

  const int tid  = threadIdx.x;
  const int lane = tid & 31;
  const int wave = tid >> 5;                          // 0..7
  const int row  = lane & 15;
  const int kb   = (lane >> 4) * 8;
  const int f    = blockIdx.z;
  const int b0   = (blockIdx.x * 8 + wave) * 16;

  const __bf16* gwr = WR + ((size_t)f << 12);
  const __bf16* gwi = WI + ((size_t)f << 12);

#if CDNA5_ASYNC_STAGE
#pragma unroll
  for (int pass = 0; pass < 2; ++pass) {              // 256 thr x 16B x 2 = 8 KB
    int e = pass * 2048 + tid * 8;
    unsigned dr = (unsigned)(uintptr_t)(lwr + e);     // low 32 bits = LDS offset
    unsigned di = (unsigned)(uintptr_t)(lwi + e);
    asm volatile("global_load_async_to_lds_b128 %0, %1, off"
                 :: "v"(dr), "v"(gwr + e) : "memory");
    asm volatile("global_load_async_to_lds_b128 %0, %1, off"
                 :: "v"(di), "v"(gwi + e) : "memory");
  }
  asm volatile("s_wait_asynccnt 0" ::: "memory");
#else
  for (int e = tid; e < KOUT * KIN; e += 256) {
    lwr[e] = gwr[e];
    lwi[e] = gwi[e];
  }
#endif
  __syncthreads();

  const __bf16* xr = Xr + ((size_t)f * BATCH + b0 + row) * KIN;
  const __bf16* xi = Xi + ((size_t)f * BATCH + b0 + row) * KIN;

  v8f accR[4] = {};
  v8f accI[4] = {};
#pragma unroll
  for (int kk = 0; kk < KIN; kk += 32) {
    v16bf ar = frag_bf16(xr, kk, kb);
    v16bf ai = frag_bf16(xi, kk, kb);
    v16bf an;                                         // -ai (bf16 NEG bits are C-only)
#pragma unroll
    for (int t = 0; t < 16; ++t) an[t] = (__bf16)(-(float)ai[t]);
#pragma unroll
    for (int it = 0; it < 4; ++it) {                  // all 4 i-tiles reuse A frags
      const __bf16* br_row = lwr + (it * 16 + row) * KIN;
      const __bf16* bi_row = lwi + (it * 16 + row) * KIN;
      v16bf br = frag_bf16(br_row, kk, kb);
      v16bf bi = frag_bf16(bi_row, kk, kb);
      accR[it] = wmma_bf16(ar, br, accR[it]);         // Yr = Xr*Wr - Xi*Wi
      accR[it] = wmma_bf16(an, bi, accR[it]);
      accI[it] = wmma_bf16(ar, bi, accI[it]);         // Yi = Xr*Wi + Xi*Wr
      accI[it] = wmma_bf16(ai, br, accI[it]);
    }
  }

  const int m = (lane >> 4) * 8, n = lane & 15;
#pragma unroll
  for (int it = 0; it < 4; ++it) {
#pragma unroll
    for (int r = 0; r < 8; ++r) {
      size_t b = (size_t)(b0 + m + r);
      size_t i = (size_t)(it * 16 + n);
      size_t o = (i * BATCH + b) * BLK + f;           // [i][b][f]
      Yr[o] = (__bf16)accR[it][r];
      Yi[o] = (__bf16)accI[it][r];
    }
  }
}

// ---- kernel 3: inverse DFT per output block, fp32 store -------------------
// Per wave: 16 batch rows x 64 output samples (4 tiles, A reused 4x), K = 256.
// out[b, i*256+n] = sum_f Yr*cos(2pi f n/256)/256 + Yi*(-sin)/256
__global__ __launch_bounds__(32) void k3_idft(const __bf16* __restrict__ Yr,
                                              const __bf16* __restrict__ Yi,
                                              const __bf16* __restrict__ icosT,
                                              const __bf16* __restrict__ isinT,
                                              float* __restrict__ out) {
  const int lane = threadIdx.x;
  const int row  = lane & 15;
  const int kb   = (lane >> 4) * 8;
  const int b0   = blockIdx.x * 16;
  const int i    = blockIdx.y;
  const int n0g  = blockIdx.z * 64;                  // 4 n-tiles per wave

  const __bf16* yr = Yr + ((size_t)i * BATCH + b0 + row) * BLK;
  const __bf16* yi = Yi + ((size_t)i * BATCH + b0 + row) * BLK;

  v8f acc[4] = {};
#pragma unroll
  for (int kk = 0; kk < BLK; kk += 32) {
    v16bf ar = frag_bf16(yr, kk, kb);                // A pair reused 4x
    v16bf ai = frag_bf16(yi, kk, kb);
#pragma unroll
    for (int nt = 0; nt < 4; ++nt) {
      const __bf16* bc = icosT + (size_t)(n0g + nt * 16 + row) * BLK;
      const __bf16* bs = isinT + (size_t)(n0g + nt * 16 + row) * BLK;
      acc[nt] = wmma_bf16(ar, frag_bf16(bc, kk, kb), acc[nt]);
      acc[nt] = wmma_bf16(ai, frag_bf16(bs, kk, kb), acc[nt]);
    }
  }

  const int m = (lane >> 4) * 8, n = lane & 15;
#pragma unroll
  for (int nt = 0; nt < 4; ++nt) {
#pragma unroll
    for (int r = 0; r < 8; ++r) {
      out[(size_t)(b0 + m + r) * INF + i * BLK + n0g + nt * 16 + n] = acc[nt][r];
    }
  }
}

// ---------------------------------------------------------------------------
extern "C" void kernel_launch(void* const* d_in, const int* in_sizes, int n_in,
                              void* d_out, int out_size, void* d_ws, size_t ws_size,
                              hipStream_t stream) {
  (void)in_sizes; (void)n_in; (void)out_size; (void)ws_size;
  const float* x      = (const float*)d_in[0];
  const float* W_real = (const float*)d_in[1];
  const float* W_imag = (const float*)d_in[2];
  float* out = (float*)d_out;
  char* ws = (char*)d_ws;

  __bf16* cosF  = (__bf16*)(ws + WS_COSF);
  __bf16* sinF  = (__bf16*)(ws + WS_SINF);
  __bf16* icosT = (__bf16*)(ws + WS_ICOS);
  __bf16* isinT = (__bf16*)(ws + WS_ISIN);
  __bf16* WR    = (__bf16*)(ws + WS_WR);
  __bf16* WI    = (__bf16*)(ws + WS_WI);
  __bf16* Xr    = (__bf16*)(ws + WS_XR);
  __bf16* Xi    = (__bf16*)(ws + WS_XI);
  __bf16* Yr    = (__bf16*)(ws + WS_YR);
  __bf16* Yi    = (__bf16*)(ws + WS_YI);

  k0_tables<<<256, 256, 0, stream>>>(cosF, sinF, icosT, isinT);
  k0_w<<<4096, 256, 0, stream>>>(W_real, W_imag, WR, WI);
  k1_fdft<<<dim3(BATCH / 16, KIN, BLK / 64), 32, 0, stream>>>(x, cosF, sinF, Xr, Xi);
  k2_fgemm<<<dim3(BATCH / 128, 1, BLK), 256, 0, stream>>>(Xr, Xi, WR, WI, Yr, Yi);
  k3_idft<<<dim3(BATCH / 16, KOUT, BLK / 64), 32, 0, stream>>>(Yr, Yi, icosT, isinT, out);
}
